// SpatialCrossAttention_56435870269920
// MI455X (gfx1250) — matrix operand order repaired and checked by reading, not compiled, not run on hardware
//
#include <hip/hip_runtime.h>

typedef __attribute__((ext_vector_type(16))) _Float16 v16h;
typedef __attribute__((ext_vector_type(8)))  _Float16 v8h;
typedef __attribute__((ext_vector_type(4)))  _Float16 v4h;
typedef __attribute__((ext_vector_type(8)))  float    v8f;

#define N_CAM  6
#define N_REF  4
#define NH     8
#define EMBD   256
#define HS     32
#define SEQL   100
#define BSZ    2
#define CIN    (N_CAM*EMBD)   // 1536
#define FEATSZ (SEQL*EMBD)    // 25600 floats per (b,cam)
#define JPAD   128            // padded key/value length for V

#define SHUF16(lo, hi) __builtin_shufflevector((lo), (hi), 0,1,2,3,4,5,6,7,8,9,10,11,12,13,14,15)

// ---------------------------------------------------------------------------
// Kernel 1 (prep):
//   Wfrag: Wsum[k][e] = sum_ref W_v[k][ref*256+e], stored B-fragment-packed:
//          Wfrag[((kt*2+half)*256 + e)*16 + (k within 16)]  (f16)
//   VbarT: planes 0..1 zero-filled (vbar_kernel overwrites the valid region);
//          plane 2 is a write-only dump for branchless tail stores.
// ---------------------------------------------------------------------------
__global__ __launch_bounds__(256) void prep_kernel(const float* __restrict__ Wv,
                                                   _Float16* __restrict__ Wfrag,
                                                   _Float16* __restrict__ VbarT) {
  int idx = blockIdx.x * 256 + threadIdx.x;
  if (idx < BSZ * EMBD * JPAD) VbarT[idx] = (_Float16)0.f;
  if (idx < CIN * EMBD) {
    int k = idx >> 8, e = idx & 255;
    const float* p = Wv + (size_t)k * (N_REF * EMBD) + e;
    float s = p[0] + p[256] + p[512] + p[768];
    int kt = k >> 5, r5 = k & 31, hf = r5 >> 4, i16 = r5 & 15;
    Wfrag[((size_t)((kt * 2 + hf) * EMBD + e) << 4) + i16] = (_Float16)s;
  }
}

// ---------------------------------------------------------------------------
// Kernel 2: VbarT[b][e][j] = (v_in(200x1536) @ Wsum(1536x256) + bsum)^T (f16)
//   one wave per 16x16 tile; 48 x wmma_f32_16x16x32_f16; fully branchless
// ---------------------------------------------------------------------------
__global__ __launch_bounds__(32) void vbar_kernel(const float* __restrict__ feat,
                                                  const _Float16* __restrict__ Wfrag,
                                                  const float* __restrict__ bv,
                                                  _Float16* __restrict__ VbarT) {
  const int mt = blockIdx.x;          // 0..12 (rows of 200, padded to 208)
  const int nt = blockIdx.y;          // 0..15 (cols of 256)
  const int lane = threadIdx.x;
  const int m = lane & 15, hf = lane >> 4;
  const int col = nt * 16 + m;

  const float bsumv = bv[col] + bv[256 + col] + bv[512 + col] + bv[768 + col];

  const int row  = mt * 16 + m;
  const float msk = (row < BSZ * SEQL) ? 1.f : 0.f;
  const int crow = (row < BSZ * SEQL) ? row : 0;   // clamp -> in-bounds loads
  const int bb   = crow >= SEQL;
  const int jj   = crow - bb * SEQL;
  const float* featrow = feat + (size_t)bb * N_CAM * FEATSZ + (size_t)jj * EMBD;

  v8f acc = {};
  for (int kt = 0; kt < CIN / 32; ++kt) {
    // A fragment: two contiguous 8-float runs (each stays inside one camera blk)
    int k0 = kt * 32 + (hf << 4);
    int k1 = k0 + 8;
    const float4* p0 = (const float4*)(featrow + (size_t)(k0 >> 8) * FEATSZ + (k0 & 255));
    const float4* p1 = (const float4*)(featrow + (size_t)(k1 >> 8) * FEATSZ + (k1 & 255));
    float4 f0 = p0[0], f1 = p0[1], f2 = p1[0], f3 = p1[1];
    v16h a;
    a[0]  = (_Float16)(f0.x * msk); a[1]  = (_Float16)(f0.y * msk);
    a[2]  = (_Float16)(f0.z * msk); a[3]  = (_Float16)(f0.w * msk);
    a[4]  = (_Float16)(f1.x * msk); a[5]  = (_Float16)(f1.y * msk);
    a[6]  = (_Float16)(f1.z * msk); a[7]  = (_Float16)(f1.w * msk);
    a[8]  = (_Float16)(f2.x * msk); a[9]  = (_Float16)(f2.y * msk);
    a[10] = (_Float16)(f2.z * msk); a[11] = (_Float16)(f2.w * msk);
    a[12] = (_Float16)(f3.x * msk); a[13] = (_Float16)(f3.y * msk);
    a[14] = (_Float16)(f3.z * msk); a[15] = (_Float16)(f3.w * msk);

    // B fragment: one contiguous 32-byte run in packed Wfrag
    const v8h* pb = (const v8h*)(Wfrag + ((size_t)((kt * 2 + hf) * EMBD + col) << 4));
    v8h blo = pb[0], bhi = pb[1];
    v16h bfr = SHUF16(blo, bhi);

    acc = __builtin_amdgcn_wmma_f32_16x16x32_f16(false, a, false, bfr,
                                                 (short)0, acc, false, false);
  }

  // Branchless vector store: each lane's 8 rows are contiguous j in VbarT.
  // 4-row runs starting at a multiple of 4 never straddle a plane boundary
  // (boundaries 100/200 are multiples of 4); rows >= 200 hit the dump plane.
  v4h lo, hi;
#pragma unroll
  for (int r = 0; r < 4; ++r) lo[r] = (_Float16)(acc[r] + bsumv);
#pragma unroll
  for (int r = 0; r < 4; ++r) hi[r] = (_Float16)(acc[4 + r] + bsumv);
  {
    int g0 = mt * 16 + (hf << 3);
    int g1 = g0 + 4;
    int b0 = (g0 >= SEQL) + (g0 >= 2 * SEQL);
    int b1 = (g1 >= SEQL) + (g1 >= 2 * SEQL);
    int j0 = g0 - b0 * SEQL;
    int j1 = g1 - b1 * SEQL;
    *(v4h*)(VbarT + ((size_t)(b0 * EMBD + col) << 7) + j0) = lo;
    *(v4h*)(VbarT + ((size_t)(b1 * EMBD + col) << 7) + j1) = hi;
  }
}

// ---------------------------------------------------------------------------
// Kernel 3: collapsed attention, one wave per (16 query rows, b, h)
// ---------------------------------------------------------------------------
__global__ __launch_bounds__(32) void attn_kernel(const float* __restrict__ feat,
                                                  const float* __restrict__ bevq,
                                                  const _Float16* __restrict__ VbarT,
                                                  float* __restrict__ out) {
  __shared__ __align__(16) float St[608][16];          // scores, key-major
  __shared__ __align__(16) _Float16 E[16][JPAD];

  const int mt = blockIdx.x;          // 0..6
  const int bh = blockIdx.y;          // 0..15
  const int bb = bh >> 3, h = bh & 7;
  const int lane = threadIdx.x;
  const int m = lane & 15, hf = lane >> 4;

  const float qscale = 0.1767766952966369f; // 1/sqrt(32)

  // Q fragment (two contiguous 8-float runs), reused across all key tiles
  v16h aq;
  {
    int i = mt * 16 + m;
    float qm = (i < SEQL) ? qscale : 0.f;
    int ic = (i < SEQL) ? i : 0;
    const float* qrow = bevq + (size_t)(bb * SEQL + ic) * EMBD + h * HS;
    const float4* q0 = (const float4*)(qrow + (hf << 4));
    const float4* q1 = (const float4*)(qrow + 8 + (hf << 4));
    float4 f0 = q0[0], f1 = q0[1], f2 = q1[0], f3 = q1[1];
    aq[0]  = (_Float16)(f0.x * qm); aq[1]  = (_Float16)(f0.y * qm);
    aq[2]  = (_Float16)(f0.z * qm); aq[3]  = (_Float16)(f0.w * qm);
    aq[4]  = (_Float16)(f1.x * qm); aq[5]  = (_Float16)(f1.y * qm);
    aq[6]  = (_Float16)(f1.z * qm); aq[7]  = (_Float16)(f1.w * qm);
    aq[8]  = (_Float16)(f2.x * qm); aq[9]  = (_Float16)(f2.y * qm);
    aq[10] = (_Float16)(f2.z * qm); aq[11] = (_Float16)(f2.w * qm);
    aq[12] = (_Float16)(f3.x * qm); aq[13] = (_Float16)(f3.y * qm);
    aq[14] = (_Float16)(f3.z * qm); aq[15] = (_Float16)(f3.w * qm);
  }

  const float* featb = feat + (size_t)bb * N_CAM * FEATSZ;

  // --- scores: 38 key tiles of 16 (600 keys padded to 608) ---
  for (int nt = 0; nt < 38; ++nt) {
    int key = nt * 16 + m;
    float km = (key < N_CAM * SEQL) ? 1.f : 0.f;
    int kc = (key < N_CAM * SEQL) ? key : 0;
    int cam = kc / SEQL;
    int jk  = kc - cam * SEQL;
    // K^T fragment: one contiguous 16-float run per lane
    const float4* kp = (const float4*)(featb + (size_t)cam * FEATSZ +
                                       (size_t)jk * EMBD + h * HS + (hf << 4));
    float4 f0 = kp[0], f1 = kp[1], f2 = kp[2], f3 = kp[3];
    v16h bfr;
    bfr[0]  = (_Float16)(f0.x * km); bfr[1]  = (_Float16)(f0.y * km);
    bfr[2]  = (_Float16)(f0.z * km); bfr[3]  = (_Float16)(f0.w * km);
    bfr[4]  = (_Float16)(f1.x * km); bfr[5]  = (_Float16)(f1.y * km);
    bfr[6]  = (_Float16)(f1.z * km); bfr[7]  = (_Float16)(f1.w * km);
    bfr[8]  = (_Float16)(f2.x * km); bfr[9]  = (_Float16)(f2.y * km);
    bfr[10] = (_Float16)(f2.z * km); bfr[11] = (_Float16)(f2.w * km);
    bfr[12] = (_Float16)(f3.x * km); bfr[13] = (_Float16)(f3.y * km);
    bfr[14] = (_Float16)(f3.z * km); bfr[15] = (_Float16)(f3.w * km);

    v8f c = {};
    c = __builtin_amdgcn_wmma_f32_16x16x32_f16(false, aq, false, bfr,
                                               (short)0, c, false, false);
    // key-major store: each lane's 8 rows are contiguous -> 2 x ds_store_b128
    float4 slo, shi;
    slo.x = c[0]; slo.y = c[1]; slo.z = c[2]; slo.w = c[3];
    shi.x = c[4]; shi.y = c[5]; shi.z = c[6]; shi.w = c[7];
    float* sp = &St[nt * 16 + m][hf << 3];
    *(float4*)(sp)     = slo;
    *(float4*)(sp + 4) = shi;
  }
  __syncthreads();

  // --- stable softmax, folding cameras; 1/(4Z) folds the ref repeat ---
  if (lane < 16) {
    int r = lane;
    int i = mt * 16 + r;
    if (i < SEQL) {
      float mx = -1e30f;
      for (int kidx = 0; kidx < N_CAM * SEQL; ++kidx) mx = fmaxf(mx, St[kidx][r]);
      float Z = 0.f;
      for (int j = 0; j < SEQL; ++j) {
        float ev = 0.f;
#pragma unroll
        for (int cam = 0; cam < N_CAM; ++cam)
          ev += __expf(St[cam * SEQL + j][r] - mx);
        Z += ev;
        St[j][r] = ev;                // slot only re-read for this j
      }
      float inv = 1.0f / (4.0f * Z);  // n_ref * Z
      for (int j = 0; j < SEQL; ++j) E[r][j] = (_Float16)(St[j][r] * inv);
      for (int j = SEQL; j < JPAD; ++j) E[r][j] = (_Float16)0.f;
    } else {
      for (int j = 0; j < JPAD; ++j) E[r][j] = (_Float16)0.f;
    }
  }
  __syncthreads();

  // --- out = E(16x128) @ VbarT_head(128x32); V pad rows pre-zeroed ---
  const _Float16* vbase = VbarT + ((size_t)(bb * EMBD + h * HS) << 7);
#pragma unroll
  for (int dn = 0; dn < 2; ++dn) {
    const _Float16* vcol = vbase + ((size_t)(dn * 16 + m) << 7);
    v8f acc = {};
#pragma unroll
    for (int kt = 0; kt < 4; ++kt) {
      // A from LDS: two aligned 16B runs of E row m
      const v8h* pe0 = (const v8h*)&E[m][kt * 32 + (hf << 4)];
      const v8h* pe1 = (const v8h*)&E[m][kt * 32 + 8 + (hf << 4)];
      v16h a = SHUF16(pe0[0], pe1[0]);
      // B from transposed V: one contiguous 32B run per lane
      const v8h* pv = (const v8h*)(vcol + kt * 32 + (hf << 4));
      v16h bfr = SHUF16(pv[0], pv[1]);
      acc = __builtin_amdgcn_wmma_f32_16x16x32_f16(false, a, false, bfr,
                                                   (short)0, acc, false, false);
    }
    // Uniform branch: only the last M-tile needs per-row guards
    float* orow = out + (size_t)(bb * SEQL) * EMBD + h * HS + dn * 16 + m;
    if (mt < 6) {
#pragma unroll
      for (int r = 0; r < 8; ++r) {
        int i = mt * 16 + r + (hf << 3);
        orow[(size_t)i * EMBD] = acc[r];
      }
    } else {
#pragma unroll
      for (int r = 0; r < 8; ++r) {
        int i = mt * 16 + r + (hf << 3);
        if (i < SEQL) orow[(size_t)i * EMBD] = acc[r];
      }
    }
  }
}

// ---------------------------------------------------------------------------
extern "C" void kernel_launch(void* const* d_in, const int* in_sizes, int n_in,
                              void* d_out, int out_size, void* d_ws, size_t ws_size,
                              hipStream_t stream) {
  const float* feat = (const float*)d_in[0];   // (2,1,6,256,10,10) == (2,6,100,256)
  const float* bevq = (const float*)d_in[1];   // (2,100,256)
  const float* Wv   = (const float*)d_in[2];   // (1536,1024)
  const float* bv   = (const float*)d_in[3];   // (1024,)
  float* out = (float*)d_out;                  // (2,100,256) f32

  _Float16* Wfrag = (_Float16*)d_ws;                    // 1536*256 halves (packed)
  _Float16* VbarT = Wfrag + (size_t)CIN * EMBD;         // 3*256*128 halves (plane 2 = dump)

  prep_kernel<<<(CIN * EMBD) / 256, 256, 0, stream>>>(Wv, Wfrag, VbarT);

  dim3 g2(13, 16);                             // 208x256 tile grid
  vbar_kernel<<<g2, 32, 0, stream>>>(feat, Wfrag, bv, VbarT);

  dim3 g3(7, 16);                              // 7 query tiles x (b,h)
  attn_kernel<<<g3, 32, 0, stream>>>(feat, bevq, VbarT, out);
}
